// LorentzLayer_38276748542434
// MI455X (gfx1250) — matrix operand order repaired and checked by reading, not compiled, not run on hardware
//
#include <hip/hip_runtime.h>
#include <math.h>

// ---------------------------------------------------------------------------
// Problem constants (from the reference)
// ---------------------------------------------------------------------------
#define D_NODES 15000
#define E_EDGES 300000
#define NROWS   (D_NODES + E_EDGES)
#define DN_ 100
#define DE_ 100
#define DT_ 100
#define KV_IN  301     // 1 + DN + DE + DT
#define KV_PAD 320     // padded to multiple of 32 for WMMA K
#define Q_IN   201     // 1 + DN + DT
#define Q_PAD  224

typedef __attribute__((ext_vector_type(16))) __bf16 v16bf;
typedef __attribute__((ext_vector_type(8)))  float  v8f;

struct __attribute__((aligned(16))) U4x { unsigned int a, b, c, d; };
struct __attribute__((aligned(16))) F4x { float a, b, c, d; };

union Frag {
    v16bf          v;
    unsigned short us[16];
    U4x            q[2];
};

// float -> bf16 bits, round-to-nearest-even
__device__ __forceinline__ unsigned short f2bf(float f) {
    unsigned u = __float_as_uint(f);
    u += 0x7FFFu + ((u >> 16) & 1u);
    return (unsigned short)(u >> 16);
}

__device__ __forceinline__ void atomicMaxFloat(float* addr, float v) {
    if (v >= 0.f) atomicMax((int*)addr, __float_as_int(v));
    else          atomicMin((unsigned int*)addr, __float_as_uint(v));
}

// Async global->LDS copy of one 16B beat per lane (CDNA5 async path,
// tracked by ASYNCcnt; bypasses VGPR staging).
__device__ __forceinline__ void async_copy_b128(unsigned lds_byte_off,
                                                const void* gaddr) {
    asm volatile("global_load_async_to_lds_b128 %0, %1, off"
                 :
                 : "v"(lds_byte_off), "v"((unsigned long long)gaddr)
                 : "memory");
}
__device__ __forceinline__ void async_wait0() {
    asm volatile("s_wait_asynccnt 0x0" ::: "memory");
}

// ---------------------------------------------------------------------------
// Lorentz-linear epilogue on WMMA accumulators.
// C layout (16x16 f32 tile): lanes 0-15 hold M=r (VGPR r), N=lane;
//                            lanes 16-31 hold M=r+8,     N=lane-16.
// ---------------------------------------------------------------------------
__device__ __forceinline__ void lorentz_post_store(
    v8f* acc, const float* __restrict__ bias, float log_s,
    float* __restrict__ out, int tileBase, int nrowsValid, int lane)
{
    const int nloc = lane & 15;
    const int half = lane >> 4;
    float bv[8];
#pragma unroll
    for (int nt = 0; nt < 8; ++nt) bv[nt] = bias[nt * 16 + nloc];
    const float esc = expf(log_s);
#pragma unroll
    for (int r = 0; r < 8; ++r) {
        float vals[8];
        float p = 0.f;
#pragma unroll
        for (int nt = 0; nt < 8; ++nt) {
            vals[nt] = acc[nt][r] + bv[nt];
            p += vals[nt] * vals[nt];
        }
        // reduce row sum across the 16-lane group (xor masks stay in-half)
#pragma unroll
        for (int msk = 1; msk < 16; msk <<= 1) p += __shfl_xor(p, msk);
        const float x0 = __shfl(vals[0], lane & 16);   // column 0 of this row
        const float sumxn = p - x0 * x0;
        const float tme = 1.f / (1.f + expf(-x0)) * esc + 1.1f;
        const float sq = sqrtf((tme * tme - 1.f) / fmaxf(sumxn, 1e-8f));
        const int row = tileBase + r + half * 8;
        if (row < nrowsValid) {
#pragma unroll
            for (int nt = 0; nt < 8; ++nt) {
                const int col = nt * 16 + nloc;
                out[(size_t)row * 128 + col] = (col == 0) ? tme : vals[nt] * sq;
            }
        }
    }
}

// ---------------------------------------------------------------------------
// Weight prep: fp32 (fout=128, fin) -> bf16 [n][kpad] (zero-padded K)
// ---------------------------------------------------------------------------
__global__ void prep_w_kernel(const float* __restrict__ W, int fin, int kpad,
                              unsigned short* __restrict__ out)
{
    const int idx = blockIdx.x * blockDim.x + threadIdx.x;
    if (idx >= 128 * kpad) return;
    const int n = idx / kpad, k = idx - n * kpad;
    out[idx] = f2bf((k < fin) ? W[(size_t)n * fin + k] : 0.f);
}

// ---------------------------------------------------------------------------
// Fused: edge features -> expmap0 -> leakyReLU -> (E x 320)x(320 x 128) WMMA
//        -> lorentz epilogue -> K_ori.   One wave = 16 edges x 128 cols.
// ---------------------------------------------------------------------------
__global__ void __launch_bounds__(128) edge_kori_kernel(
    const float* __restrict__ h,  const float* __restrict__ ef,
    const float* __restrict__ dt, const float* __restrict__ te_w,
    const float* __restrict__ te_b, const unsigned short* __restrict__ Wkv,
    const float* __restrict__ bias, const float* __restrict__ log_s_p,
    float* __restrict__ Kori)
{
    __shared__ __align__(16) unsigned short feat[4 * 16 * KV_PAD];  // 40 KB
    const int lane = threadIdx.x & 31;
    const int wave = threadIdx.x >> 5;
    const int tileBase = (blockIdx.x * 4 + wave) * 16;
    unsigned short* slab = &feat[wave * 16 * KV_PAD];

    for (int r = 0; r < 16; ++r) {
        const int e = tileBase + r;
        float spv[10];
        float ss = 0.f;
#pragma unroll
        for (int it = 0; it < 10; ++it) {
            const int j = lane + it * 32;
            float v = 0.f;
            if (e < E_EDGES && j < 300) {
                if (j < 100)      v = h[(size_t)(D_NODES + e) * DN_ + j];
                else if (j < 200) v = ef[(size_t)e * DE_ + (j - 100)];
                else {
                    const int t = j - 200;
                    v = cosf(dt[e] * te_w[t] + te_b[t]);
                }
            }
            spv[it] = v;
            ss += v * v;
        }
#pragma unroll
        for (int msk = 16; msk >= 1; msk >>= 1) ss += __shfl_xor(ss, msk);
        const float nrm = sqrtf(fmaxf(ss, 1e-8f));
        const float ch = coshf(nrm);
        const float sh = sinhf(nrm) / nrm;
        unsigned short* row = &slab[r * KV_PAD];
        if (lane == 0) row[0] = f2bf(ch);           // cosh >= 1, leaky no-op
#pragma unroll
        for (int it = 0; it < 10; ++it) {
            const int j = lane + it * 32;
            if (j < 300) {
                float x = sh * spv[it];
                x = (x >= 0.f) ? x : 0.2f * x;       // leaky ReLU
                row[1 + j] = f2bf(x);
            }
        }
        if (lane < KV_PAD - KV_IN) row[KV_IN + lane] = 0;  // zero pad
    }
    __syncthreads();

    v8f acc[8] = {};
    const int mrow = lane & 15;
    const int half = lane >> 4;
    for (int ks = 0; ks < KV_PAD / 32; ++ks) {
        const int koff = ks * 32;
        Frag a;                                  // ISA 16-bit A layout
        const int abase = koff + half * 8;
        a.q[0] = *(const U4x*)&slab[mrow * KV_PAD + abase];
        a.q[1] = *(const U4x*)&slab[mrow * KV_PAD + abase + 16];
        const int kb = koff + half * 8;
        Frag bfr[8];                             // decouple loads from WMMA
#pragma unroll
        for (int nt = 0; nt < 8; ++nt) {
            const int n = nt * 16 + mrow;
            bfr[nt].q[0] = *(const U4x*)&Wkv[n * KV_PAD + kb];
            bfr[nt].q[1] = *(const U4x*)&Wkv[n * KV_PAD + kb + 16];
        }
#pragma unroll
        for (int nt = 0; nt < 8; ++nt)
            acc[nt] = __builtin_amdgcn_wmma_f32_16x16x32_bf16(
                false, a.v, false, bfr[nt].v, (short)0, acc[nt], false, false);
    }
    lorentz_post_store(acc, bias, log_s_p[0], Kori, tileBase, E_EDGES, lane);
}

// ---------------------------------------------------------------------------
// Fused node path: Q features (201-dim) -> expmap0 -> leaky -> WMMA -> Q_ori
// ---------------------------------------------------------------------------
__global__ void __launch_bounds__(128) node_qori_kernel(
    const float* __restrict__ h, const float* __restrict__ te_b,
    const unsigned short* __restrict__ Wq, const float* __restrict__ bias,
    const float* __restrict__ log_s_p, float* __restrict__ Qori)
{
    __shared__ __align__(16) unsigned short feat[4 * 16 * Q_PAD];   // 28 KB
    const int lane = threadIdx.x & 31;
    const int wave = threadIdx.x >> 5;
    const int tileBase = (blockIdx.x * 4 + wave) * 16;
    unsigned short* slab = &feat[wave * 16 * Q_PAD];

    for (int r = 0; r < 16; ++r) {
        const int d = tileBase + r;
        float spv[7];
        float ss = 0.f;
#pragma unroll
        for (int it = 0; it < 7; ++it) {
            const int j = lane + it * 32;
            float v = 0.f;
            if (d < D_NODES && j < 200) {
                v = (j < 100) ? h[(size_t)d * DN_ + j] : cosf(te_b[j - 100]);
            }
            spv[it] = v;
            ss += v * v;
        }
#pragma unroll
        for (int msk = 16; msk >= 1; msk >>= 1) ss += __shfl_xor(ss, msk);
        const float nrm = sqrtf(fmaxf(ss, 1e-8f));
        const float ch = coshf(nrm);
        const float sh = sinhf(nrm) / nrm;
        unsigned short* row = &slab[r * Q_PAD];
        if (lane == 0) row[0] = f2bf(ch);
#pragma unroll
        for (int it = 0; it < 7; ++it) {
            const int j = lane + it * 32;
            if (j < 200) {
                float x = sh * spv[it];
                x = (x >= 0.f) ? x : 0.2f * x;
                row[1 + j] = f2bf(x);
            }
        }
        if (lane < Q_PAD - Q_IN) row[Q_IN + lane] = 0;
    }
    __syncthreads();

    v8f acc[8] = {};
    const int mrow = lane & 15;
    const int half = lane >> 4;
    for (int ks = 0; ks < Q_PAD / 32; ++ks) {
        const int koff = ks * 32;
        Frag a;
        const int abase = koff + half * 8;
        a.q[0] = *(const U4x*)&slab[mrow * Q_PAD + abase];
        a.q[1] = *(const U4x*)&slab[mrow * Q_PAD + abase + 16];
        const int kb = koff + half * 8;
        Frag bfr[8];
#pragma unroll
        for (int nt = 0; nt < 8; ++nt) {
            const int n = nt * 16 + mrow;
            bfr[nt].q[0] = *(const U4x*)&Wq[n * Q_PAD + kb];
            bfr[nt].q[1] = *(const U4x*)&Wq[n * Q_PAD + kb + 16];
        }
#pragma unroll
        for (int nt = 0; nt < 8; ++nt)
            acc[nt] = __builtin_amdgcn_wmma_f32_16x16x32_bf16(
                false, a.v, false, bfr[nt].v, (short)0, acc[nt], false, false);
    }
    lorentz_post_store(acc, bias, log_s_p[0], Qori, tileBase, D_NODES, lane);
}

// ---------------------------------------------------------------------------
// Generic 315k x 128 @ 128x128 lorentz_linear.  The 32KB bf16 weight matrix
// is staged into LDS once per block via async global->LDS (ASYNCcnt), then
// all 4 waves stream A rows (fp32 -> bf16) and run WMMA against LDS B frags.
// ---------------------------------------------------------------------------
__global__ void __launch_bounds__(128) gemm128_lorentz_kernel(
    const float* __restrict__ src0, const float* __restrict__ src1,
    const int* __restrict__ gather, const unsigned short* __restrict__ Wb,
    const float* __restrict__ bias, const float* __restrict__ log_s_p,
    float* __restrict__ out, int nrows)
{
    __shared__ __align__(16) unsigned short wlds[128 * 128];   // 32 KB
    const int lane = threadIdx.x & 31;
    const int wave = threadIdx.x >> 5;

    // ---- async stage weights: 2048 x 16B beats, 16 per thread ----
    for (int t = threadIdx.x; t < 2048; t += 128) {
        const unsigned loff = (unsigned)(unsigned long long)(&wlds[t * 8]);
        async_copy_b128(loff, (const void*)(Wb + t * 8));
    }
    async_wait0();
    __syncthreads();

    const int tileBase = (blockIdx.x * 4 + wave) * 16;
    const int mrow = lane & 15;
    const int half = lane >> 4;
    const int i = tileBase + mrow;
    const bool valid = (i < nrows);
    const float* rowp = src0;
    if (valid) {
        if (i < E_EDGES) rowp = gather ? src0 + (size_t)gather[i] * 128
                                       : src0 + (size_t)i * 128;
        else             rowp = src1 + (size_t)(i - E_EDGES) * 128;
    }
    v8f acc[8] = {};
    for (int ks = 0; ks < 4; ++ks) {
        const int koff = ks * 32;
        Frag a;
        const int abase = koff + half * 8;
        if (valid) {
            F4x f0 = *(const F4x*)(rowp + abase);
            F4x f1 = *(const F4x*)(rowp + abase + 4);
            F4x f2 = *(const F4x*)(rowp + abase + 16);
            F4x f3 = *(const F4x*)(rowp + abase + 20);
            float t[16] = { f0.a, f0.b, f0.c, f0.d, f1.a, f1.b, f1.c, f1.d,
                            f2.a, f2.b, f2.c, f2.d, f3.a, f3.b, f3.c, f3.d };
#pragma unroll
            for (int j = 0; j < 16; ++j) a.us[j] = f2bf(t[j]);
        } else {
#pragma unroll
            for (int j = 0; j < 16; ++j) a.us[j] = 0;
        }
        const int kb = koff + half * 8;
        Frag bfr[8];
#pragma unroll
        for (int nt = 0; nt < 8; ++nt) {
            const int n = nt * 16 + mrow;
            bfr[nt].q[0] = *(const U4x*)&wlds[n * 128 + kb];
            bfr[nt].q[1] = *(const U4x*)&wlds[n * 128 + kb + 16];
        }
#pragma unroll
        for (int nt = 0; nt < 8; ++nt)
            acc[nt] = __builtin_amdgcn_wmma_f32_16x16x32_bf16(
                false, a.v, false, bfr[nt].v, (short)0, acc[nt], false, false);
    }
    lorentz_post_store(acc, bias, log_s_p[0], out, tileBase, nrows, lane);
}

// ---------------------------------------------------------------------------
// Segment softmax + aggregation
// ---------------------------------------------------------------------------
__global__ void init_reduce_kernel(float* __restrict__ m, float* __restrict__ s,
                                   float* __restrict__ rst)
{
    const int idx = blockIdx.x * blockDim.x + threadIdx.x;
    if (idx < D_NODES) { m[idx] = -__builtin_inff(); s[idx] = 0.f; }
    if (idx < D_NODES * 128) rst[idx] = 0.f;
}

__global__ void logits_max_kernel(const float* __restrict__ Qh,
                                  const float* __restrict__ Kh,
                                  const int* __restrict__ edge_dst,
                                  const float* __restrict__ att_scale_p,
                                  float* __restrict__ logit,
                                  float* __restrict__ mbuf)
{
    const int i = blockIdx.x * blockDim.x + threadIdx.x;
    if (i >= NROWS) return;
    const F4x* q4 = (const F4x*)(Qh + (size_t)i * 128);
    const F4x* k4 = (const F4x*)(Kh + (size_t)i * 128);
    float inner = 0.f;
#pragma unroll 4
    for (int n = 0; n < 32; ++n) {
        F4x qa = q4[n], ka = k4[n];
        inner += qa.a * ka.a + qa.b * ka.b + qa.c * ka.c + qa.d * ka.d;
    }
    inner -= 2.f * Qh[(size_t)i * 128] * Kh[(size_t)i * 128];  // minkowski
    const float lg = (2.f + 2.f * inner) / att_scale_p[0];
    logit[i] = lg;
    const int g = (i < E_EDGES) ? edge_dst[i] : (i - E_EDGES);
    atomicMaxFloat(&mbuf[g], lg);
}

__global__ void expsum_kernel(const int* __restrict__ edge_dst,
                              const float* __restrict__ mbuf,
                              float* __restrict__ logit,
                              float* __restrict__ sbuf)
{
    const int i = blockIdx.x * blockDim.x + threadIdx.x;
    if (i >= NROWS) return;
    const int g = (i < E_EDGES) ? edge_dst[i] : (i - E_EDGES);
    const float e = expf(logit[i] - mbuf[g]);
    logit[i] = e;
    atomicAdd(&sbuf[g], e);
}

__global__ void accum_kernel(const int* __restrict__ edge_dst,
                             const float* __restrict__ logit,
                             const float* __restrict__ sbuf,
                             const float* __restrict__ Vh,
                             float* __restrict__ rst)
{
    const size_t total = (size_t)NROWS * 128;
    const size_t stride = (size_t)gridDim.x * blockDim.x;
    for (size_t idx = (size_t)blockIdx.x * blockDim.x + threadIdx.x;
         idx < total; idx += stride) {
        const int i = (int)(idx >> 7);
        const int n = (int)(idx & 127);
        const int g = (i < E_EDGES) ? edge_dst[i] : (i - E_EDGES);
        const float attn = logit[i] / sbuf[g];
        atomicAdd(&rst[(size_t)g * 128 + n], Vh[idx] * attn);
    }
}

__global__ void final_kernel(const float* __restrict__ rst,
                             float* __restrict__ out)
{
    const int d = blockIdx.x * blockDim.x + threadIdx.x;
    if (d >= D_NODES) return;
    const float* r = rst + (size_t)d * 128;
    float r0 = r[0];
    float sum = 0.f;
    for (int n = 1; n < 128; ++n) sum += r[n] * r[n];
    const float inner = sum - r0 * r0;                 // minkowski(rst,rst)
    const float denom = sqrtf(fmaxf(fabsf(-inner), 1e-8f));
    const float inv = 1.f / denom;
    const float alpha = fmaxf(r0 * inv, 1.f + 1e-7f);
    const float coef = acoshf(alpha) / sqrtf(fmaxf(alpha * alpha - 1.f, 1e-8f));
    out[(size_t)d * 128] = 0.f;
    for (int n = 1; n < 128; ++n) out[(size_t)d * 128 + n] = coef * (r[n] * inv);
}

// ---------------------------------------------------------------------------
// Host orchestration
// ---------------------------------------------------------------------------
extern "C" void kernel_launch(void* const* d_in, const int* in_sizes, int n_in,
                              void* d_out, int out_size, void* d_ws, size_t ws_size,
                              hipStream_t stream)
{
    (void)in_sizes; (void)n_in; (void)out_size; (void)ws_size;
    const float* h       = (const float*)d_in[0];
    const float* ef      = (const float*)d_in[1];
    const float* dt      = (const float*)d_in[2];
    const float* te_w    = (const float*)d_in[3];
    const float* te_b    = (const float*)d_in[4];
    const float* lq_W    = (const float*)d_in[5];
    const float* lq_b    = (const float*)d_in[6];
    const float* lq_s    = (const float*)d_in[7];
    const float* lkv_W   = (const float*)d_in[8];
    const float* lkv_b   = (const float*)d_in[9];
    const float* lkv_s   = (const float*)d_in[10];
    const float* wq_W    = (const float*)d_in[11];
    const float* wq_b    = (const float*)d_in[12];
    const float* wq_s    = (const float*)d_in[13];
    const float* wk_W    = (const float*)d_in[14];
    const float* wk_b    = (const float*)d_in[15];
    const float* wk_s    = (const float*)d_in[16];
    const float* wv_W    = (const float*)d_in[17];
    const float* wv_b    = (const float*)d_in[18];
    const float* wv_s    = (const float*)d_in[19];
    const float* att_sc  = (const float*)d_in[20];
    const int*   edge_dst = (const int*)d_in[21];
    float* out = (float*)d_out;

    // workspace carve (256B aligned)
    size_t off = 0;
    char* base = (char*)d_ws;
    auto carve = [&](size_t bytes) -> char* {
        off = (off + 255) & ~(size_t)255;
        char* p = base + off;
        off += bytes;
        return p;
    };
    unsigned short* Wq_b16  = (unsigned short*)carve((size_t)Q_PAD  * 128 * 2);
    unsigned short* Wkv_b16 = (unsigned short*)carve((size_t)KV_PAD * 128 * 2);
    unsigned short* Wwq_b16 = (unsigned short*)carve((size_t)128 * 128 * 2);
    unsigned short* Wwk_b16 = (unsigned short*)carve((size_t)128 * 128 * 2);
    unsigned short* Wwv_b16 = (unsigned short*)carve((size_t)128 * 128 * 2);
    float* Qori  = (float*)carve((size_t)D_NODES * 128 * 4);
    float* Kori  = (float*)carve((size_t)E_EDGES * 128 * 4);
    float* Qh    = (float*)carve((size_t)NROWS * 128 * 4);
    float* Kh    = (float*)carve((size_t)NROWS * 128 * 4);
    float* Vh    = (float*)carve((size_t)NROWS * 128 * 4);
    float* logit = (float*)carve((size_t)NROWS * 4);
    float* mbuf  = (float*)carve((size_t)D_NODES * 4);
    float* sbuf  = (float*)carve((size_t)D_NODES * 4);
    float* rst   = (float*)carve((size_t)D_NODES * 128 * 4);

    // 1) weights -> bf16 padded
    prep_w_kernel<<<(128 * Q_PAD  + 255) / 256, 256, 0, stream>>>(lq_W,  Q_IN,  Q_PAD,  Wq_b16);
    prep_w_kernel<<<(128 * KV_PAD + 255) / 256, 256, 0, stream>>>(lkv_W, KV_IN, KV_PAD, Wkv_b16);
    prep_w_kernel<<<(128 * 128 + 255) / 256, 256, 0, stream>>>(wq_W, 128, 128, Wwq_b16);
    prep_w_kernel<<<(128 * 128 + 255) / 256, 256, 0, stream>>>(wk_W, 128, 128, Wwk_b16);
    prep_w_kernel<<<(128 * 128 + 255) / 256, 256, 0, stream>>>(wv_W, 128, 128, Wwv_b16);

    // 2) fused feature + first lorentz_linear (WMMA)
    edge_kori_kernel<<<(E_EDGES + 63) / 64, 128, 0, stream>>>(
        h, ef, dt, te_w, te_b, Wkv_b16, lkv_b, lkv_s, Kori);
    node_qori_kernel<<<(D_NODES + 63) / 64, 128, 0, stream>>>(
        h, te_b, Wq_b16, lq_b, lq_s, Qori);

    // 3) Qh / Kh / Vh (WMMA 128x128 with async-LDS weight staging)
    const int gblocks = (NROWS + 63) / 64;
    gemm128_lorentz_kernel<<<gblocks, 128, 0, stream>>>(
        Qori, Qori, edge_dst, Wwq_b16, wq_b, wq_s, Qh, NROWS);
    gemm128_lorentz_kernel<<<gblocks, 128, 0, stream>>>(
        Kori, Qori, nullptr, Wwk_b16, wk_b, wk_s, Kh, NROWS);
    gemm128_lorentz_kernel<<<gblocks, 128, 0, stream>>>(
        Kh, Kh + (size_t)E_EDGES * 128, nullptr, Wwv_b16, wv_b, wv_s, Vh, NROWS);

    // 4) segment softmax + aggregation
    init_reduce_kernel<<<(D_NODES * 128 + 255) / 256, 256, 0, stream>>>(mbuf, sbuf, rst);
    logits_max_kernel<<<(NROWS + 255) / 256, 256, 0, stream>>>(
        Qh, Kh, edge_dst, att_sc, logit, mbuf);
    expsum_kernel<<<(NROWS + 255) / 256, 256, 0, stream>>>(edge_dst, mbuf, logit, sbuf);
    accum_kernel<<<4096, 256, 0, stream>>>(edge_dst, logit, sbuf, Vh, rst);

    // 5) normalize + logmap0
    final_kernel<<<(D_NODES + 255) / 256, 256, 0, stream>>>(rst, out);
}
